// RnnModelSimple_17549236372249
// MI455X (gfx1250) — compile-verified, matrix operand order checked by reading, not run on hardware
//
#include <hip/hip_runtime.h>
#include <hip/hip_bf16.h>

#define VOCAB 10000
#define HID   256
#define BATCH 32
#define SEQ   512

typedef __attribute__((ext_vector_type(16))) _Float16 v16h;
typedef __attribute__((ext_vector_type(8)))  _Float16 v8h;
typedef __attribute__((ext_vector_type(8)))  float    v8f;
typedef __attribute__((ext_vector_type(4)))  int      v4i;

#define AS(n) __attribute__((address_space(n)))

#if defined(__AMDGCN__) && __has_builtin(__builtin_amdgcn_global_load_async_to_lds_b128)
#define HAS_ASYNC_LDS 1
#else
#define HAS_ASYNC_LDS 0
#endif

#if HAS_ASYNC_LDS
static __device__ __forceinline__ void wait_async0() {
#if __has_builtin(__builtin_amdgcn_s_wait_asynccnt)
    __builtin_amdgcn_s_wait_asynccnt(0);
#else
    asm volatile("s_wait_asynccnt 0" ::: "memory");
#endif
}
#endif

// Load a 16-half WMMA A/B fragment for one lane. Per the CDNA5 16-bit A-matrix
// layout (ISA 7.12.2), each lane needs two contiguous 8-half chunks at
// row + kbase and row + kbase + 16 (kbase already includes the +8 offset for
// lanes 16..31). Both chunks are 16B aligned -> two b128 loads.
static __device__ __forceinline__ v16h load_frag(const _Float16* __restrict__ p) {
    v8h lo = *(const v8h*)(p);
    v8h hi = *(const v8h*)(p + 16);
    v16h r;
#pragma unroll
    for (int i = 0; i < 8; ++i) { r[i] = lo[i]; r[i + 8] = hi[i]; }
    return r;
}

// ---------------------------------------------------------------------------
// Kernel A: Wd [HID, VOCAB] f32  ->  WdT [VOCAB, HID] f16 (K-contiguous rows)
// ---------------------------------------------------------------------------
__global__ void convert_wd_kernel(const float* __restrict__ Wd,
                                  _Float16* __restrict__ WdT) {
    const int gid = blockIdx.x * blockDim.x + threadIdx.x;   // [0, VOCAB*HID/4)
    const int n  = gid >> 6;            // vocab column
    const int k4 = (gid & 63) << 2;     // hid index * 4
#pragma unroll
    for (int i = 0; i < 4; ++i)
        WdT[(size_t)n * HID + k4 + i] = (_Float16)Wd[(size_t)(k4 + i) * VOCAB + n];
}

// ---------------------------------------------------------------------------
// Kernel B: persistent single-workgroup recurrence.
//   h_new = tanh(Wxh[token] + h @ Whh + bh), 512 serial steps.
// Whh lives in LDS (transposed f16, 128KB of the 320KB WGP pool) so the serial
// loop never touches HBM for weights. 8 waves each own 4 of the 32 16x16
// output tiles of the [32 x 256] state; K=256 -> 8 WMMAs per tile per step.
// Epilogue fuses the one-hot gather (row of Wxh), bias and tanh, then streams
// y_t to the workspace in f16 for the output GEMM.
// ---------------------------------------------------------------------------
__global__ void rnn_recurrence_kernel(const int*   __restrict__ inputs,  // [BATCH, SEQ]
                                      const float* __restrict__ state,   // [BATCH, HID]
                                      const float* __restrict__ Wxh,     // [VOCAB, HID]
                                      const float* __restrict__ Whh,     // [HID, HID]
                                      const float* __restrict__ bh,      // [HID]
                                      _Float16*    __restrict__ Y16,     // [SEQ*BATCH, HID]
                                      float*       __restrict__ h_final) // [BATCH, HID]
{
    extern __shared__ char smem[];
    _Float16* whhT = (_Float16*)smem;                  // [HID(n)][HID(k)]  128 KB
    _Float16* h16  = whhT + HID * HID;                 // [BATCH][HID]       16 KB
    float*    hf32 = (float*)(h16 + BATCH * HID);      // [BATCH][HID]       32 KB
    float*    bhs  = hf32 + BATCH * HID;               // [HID]               1 KB

    const int tid  = threadIdx.x;
    const int lane = tid & 31;
    const int wave = tid >> 5;

    // Stage Whh transposed (f16) into LDS; reads are coalesced along n.
    for (int i = tid; i < HID * HID; i += 256) {
        const int k = i / HID, n = i % HID;
        whhT[n * HID + k] = (_Float16)Whh[i];
    }
    for (int i = tid; i < BATCH * HID; i += 256) {
        const float v = state[i];
        hf32[i] = v;
        h16[i]  = (_Float16)v;
    }
    if (tid < HID) bhs[tid] = bh[tid];
    __syncthreads();

    const int tileM  = wave & 1;                 // 2 M-tiles cover BATCH=32
    const int tileN0 = (wave >> 1) * 4;          // 4 consecutive N-tiles per wave
    const int mrow   = tileM * 16 + (lane & 15); // A-fragment row for this lane
    const int koff   = (lane >= 16) ? 8 : 0;     // K offset for upper half-wave
    const int rbase  = (lane >= 16) ? 8 : 0;     // C/D row offset for upper half

    for (int t = 0; t < SEQ; ++t) {
        v8f acc[4] = {};
#pragma unroll
        for (int kk = 0; kk < HID / 32; ++kk) {
            const int kbase = kk * 32 + koff;
            const v16h a = load_frag(&h16[mrow * HID + kbase]);   // reused 4x
#pragma unroll
            for (int j = 0; j < 4; ++j) {
                const int n = (tileN0 + j) * 16 + (lane & 15);
                const v16h b = load_frag(&whhT[n * HID + kbase]);
                acc[j] = __builtin_amdgcn_wmma_f32_16x16x32_f16(
                    false, a, false, b, (short)0, acc[j], false, false);
            }
        }
        __syncthreads();   // all reads of old h16 complete before overwrite

#pragma unroll
        for (int j = 0; j < 4; ++j) {
            const int   n    = (tileN0 + j) * 16 + (lane & 15);
            const float bias = bhs[n];
#pragma unroll
            for (int r = 0; r < 8; ++r) {
                const int   m   = tileM * 16 + r + rbase;
                const int   tok = inputs[m * SEQ + t];
                // one-hot(x) @ Wxh == row gather from Wxh (hot in L2: 10.2 MB)
                const float v   = tanhf(acc[j][r] + Wxh[(size_t)tok * HID + n] + bias);
                h16[m * HID + n]  = (_Float16)v;
                hf32[m * HID + n] = v;
                Y16[((size_t)t * BATCH + m) * HID + n] = (_Float16)v;
            }
        }
        __syncthreads();   // new h16 visible before next step's WMMAs
    }

    for (int i = tid; i < BATCH * HID; i += 256) h_final[i] = hf32[i];
}

// ---------------------------------------------------------------------------
// Kernel C: output = Y16 [16384 x 256] @ WdT^T [256 x 10000] + bd  (f32 out).
// Block tile = 128(M) x 80(N): 8 waves stacked along M share one 40KB B panel
// staged in LDS (via GLOBAL_LOAD_ASYNC_TO_LDS_B128 where available), cutting
// B-panel L2 traffic 8x (5.1GB -> 640MB). Each wave computes a 16x80 strip:
// 5 accumulator tiles sharing each A fragment (A panels stay L2-resident:
// Y16=8MB + WdT=5MB << 192MB L2). The 655MB f32 output is written with
// non-temporal stores so the result stream doesn't evict the operand panels.
// ---------------------------------------------------------------------------
__global__ void out_gemm_kernel(const _Float16* __restrict__ Y16,  // [16384, 256]
                                const _Float16* __restrict__ WdT,  // [10000, 256]
                                const float*    __restrict__ bd,   // [10000]
                                float*          __restrict__ out)  // [16384, 10000]
{
    __shared__ __align__(16) _Float16 bpanel[80 * HID];   // 40 KB

    const int tid    = threadIdx.x;
    const int lane   = tid & 31;
    const int wave   = tid >> 5;
    const int mblock = blockIdx.x / 125;        // [0,128): 128 output rows each
    const int N0     = (blockIdx.x % 125) * 80; // 125 N-strips of 80 cols
    const int koff   = (lane >= 16) ? 8 : 0;
    const int rbase  = (lane >= 16) ? 8 : 0;

    // ---- stage B panel: WdT rows [N0, N0+80) -> LDS (40960B, 16B per thread/iter)
    {
        const _Float16* gsrc = &WdT[(size_t)N0 * HID];
#if HAS_ASYNC_LDS
#pragma unroll
        for (int i = tid; i < 80 * HID / 8; i += 256) {
            __builtin_amdgcn_global_load_async_to_lds_b128(
                (AS(1) v4i*)(gsrc + i * 8),
                (AS(3) v4i*)(&bpanel[i * 8]),
                0, 0);
        }
        wait_async0();
#else
        for (int i = tid; i < 80 * HID / 8; i += 256)
            *(v8h*)&bpanel[i * 8] = *(const v8h*)(gsrc + i * 8);
#endif
        __syncthreads();
    }

    const int mtile = mblock * 8 + wave;        // waves stacked along M
    const _Float16* arow = &Y16[(size_t)(mtile * 16 + (lane & 15)) * HID];
    __builtin_prefetch(arow, 0, 1);

    // Bias-initialized accumulators: every lane/VGPR of tile j holds column
    // N0 + j*16 + (lane&15), identical across the 8 M rows it covers.
    v8f acc[5];
#pragma unroll
    for (int j = 0; j < 5; ++j) {
        const float b = bd[N0 + j * 16 + (lane & 15)];
#pragma unroll
        for (int r = 0; r < 8; ++r) acc[j][r] = b;
    }

#pragma unroll
    for (int kk = 0; kk < HID / 32; ++kk) {
        const int kbase = kk * 32 + koff;
        const v16h a = load_frag(arow + kbase);              // reused 5x
#pragma unroll
        for (int j = 0; j < 5; ++j) {
            const int nloc = j * 16 + (lane & 15);           // [0,80)
            const v16h b = load_frag(&bpanel[nloc * HID + kbase]);
            acc[j] = __builtin_amdgcn_wmma_f32_16x16x32_f16(
                false, a, false, b, (short)0, acc[j], false, false);
        }
    }

#pragma unroll
    for (int j = 0; j < 5; ++j) {
        const int n = N0 + j * 16 + (lane & 15);
#pragma unroll
        for (int r = 0; r < 8; ++r) {
            const int row = mtile * 16 + r + rbase;
            __builtin_nontemporal_store(acc[j][r], &out[(size_t)row * VOCAB + n]);
        }
    }
}

extern "C" void kernel_launch(void* const* d_in, const int* in_sizes, int n_in,
                              void* d_out, int out_size, void* d_ws, size_t ws_size,
                              hipStream_t stream) {
    const int*   inputs = (const int*)  d_in[0];  // [BATCH, SEQ] int32
    const float* state  = (const float*)d_in[1];  // [BATCH, HID]
    const float* Wxh    = (const float*)d_in[2];  // [VOCAB, HID]
    const float* Whh    = (const float*)d_in[3];  // [HID, HID]
    const float* bh     = (const float*)d_in[4];  // [HID]
    const float* Wd     = (const float*)d_in[5];  // [HID, VOCAB]
    const float* bd     = (const float*)d_in[6];  // [VOCAB]

    float* out     = (float*)d_out;                          // [SEQ*BATCH, VOCAB]
    float* h_final = out + (size_t)SEQ * BATCH * VOCAB;      // [BATCH, HID]

    _Float16* Y16 = (_Float16*)d_ws;                         // 8 MB
    _Float16* WdT = Y16 + (size_t)SEQ * BATCH * HID;         // 5 MB

    // 1) Wd -> f16, transposed to K-contiguous rows (independent of recurrence)
    convert_wd_kernel<<<(VOCAB * HID / 4) / 256, 256, 0, stream>>>(Wd, WdT);

    // 2) persistent single-WGP recurrence; Whh/h resident in 181 KB of LDS
    const size_t smem = (size_t)HID * HID * 2    // whhT f16
                      + (size_t)BATCH * HID * 2  // h16
                      + (size_t)BATCH * HID * 4  // hf32
                      + (size_t)HID * 4;         // bh
    rnn_recurrence_kernel<<<1, 256, smem, stream>>>(inputs, state, Wxh, Whh, bh,
                                                    Y16, h_final);

    // 3) big output GEMM: 128x80 block tiles, B panel staged in LDS
    out_gemm_kernel<<<16000, 256, 0, stream>>>(Y16, WdT, bd, out);
}
